// MambaMixer_56727928046135
// MI455X (gfx1250) — matrix operand order repaired and checked by reading, not compiled, not run on hardware
//
#include <hip/hip_runtime.h>
#include <hip/hip_bf16.h>
#include <math.h>

// ---- problem dims (fixed by reference) ----
#define Bn   2
#define Ln   2048
#define DMn  1024
#define DIn  2048
#define DSn  16
#define DCn  4
#define DTRn 64
#define Mrows (Bn*Ln)          // 4096
#define NXZ   (2*DIn)          // 4096
#define NXP   128              // x_proj N padded 96 -> 128

typedef __attribute__((ext_vector_type(16))) __bf16 v16bf;
typedef __attribute__((ext_vector_type(8)))  __bf16 v8bf;
typedef __attribute__((ext_vector_type(8)))  float  v8f;
typedef int v4ig __attribute__((vector_size(16)));   // GCC-vector int4 (builtin pointee type)

#define AS1 __attribute__((address_space(1)))
#define AS3 __attribute__((address_space(3)))

#if __has_builtin(__builtin_amdgcn_global_load_async_to_lds_b128)
#define USE_ASYNC 1
#else
#define USE_ASYNC 0
#endif

__device__ __forceinline__ void async_b128(const void* g, void* l)
{
#if USE_ASYNC
    __builtin_amdgcn_global_load_async_to_lds_b128(
        (AS1 v4ig*)(uintptr_t)g, (AS3 v4ig*)(uintptr_t)l, 0, 0);
#else
    (void)g; (void)l;
#endif
}

__device__ __forceinline__ void wait_async0()
{
#if __has_builtin(__builtin_amdgcn_s_wait_asynccnt)
    __builtin_amdgcn_s_wait_asynccnt(0);
#else
    asm volatile("s_wait_asynccnt 0x0" ::: "memory");
#endif
}

// ============================================================================
// bf16 WMMA GEMM:  C[M,N] = A[M,K] * Bt[K,N]   (A,Bt bf16 row-major, C fp32)
// block = 256 threads = 8 waves (2x4), block tile 64(M) x 128(N),
// wave tile 32x32 via 2x2 fragments, K-step 32 (v_wmma_f32_16x16x32_bf16).
// A/B tiles staged through LDS with double-buffered async global->LDS loads.
// All dims must be multiples of the tile (buffers are padded to guarantee it).
// ============================================================================
#define ASTR 40    // 32 + 8 bf16 pad (80B row, multiple of 16B)
#define BSTR 136   // 128 + 8 bf16 pad (272B row, multiple of 16B)

__global__ __launch_bounds__(256) void gemm_bf16_wmma(
    const __bf16* __restrict__ A, const __bf16* __restrict__ Bt,
    float* __restrict__ C, int M, int N, int K)
{
    __shared__ __align__(16) __bf16 As[2][64][ASTR];   // 10 KB
    __shared__ __align__(16) __bf16 Bs[2][32][BSTR];   // 17 KB

    const int t    = threadIdx.x;
    const int wave = t >> 5;
    const int lane = t & 31;
    const int wm = wave >> 2;          // 0..1
    const int wn = wave & 3;           // 0..3
    const int mBlk = blockIdx.y * 64;
    const int nBlk = blockIdx.x * 128;

    const int hb = lane >> 4;          // half-wave select
    const int r  = lane & 15;

    // staging assignment: A 64x32 (16B/thread), B 32x128 (32B/thread)
    const int ar = t >> 2;             // 0..63
    const int ac = (t & 3) * 8;        // 0,8,16,24
    const int br = t >> 3;             // 0..31
    const int bc = (t & 7) * 16;       // 0..112

    v8f acc[2][2];
#pragma unroll
    for (int i = 0; i < 2; ++i)
#pragma unroll
        for (int j = 0; j < 2; ++j)
#pragma unroll
            for (int tt = 0; tt < 8; ++tt) acc[i][j][tt] = 0.0f;

    const int KT = K >> 5;
    int buf = 0;

#if USE_ASYNC
    // prologue: stage k=0 tile into buffer 0
    {
        const __bf16* ga = A + (size_t)(mBlk + ar) * K + ac;
        async_b128(ga, &As[0][ar][ac]);
        const __bf16* gb = Bt + (size_t)br * N + nBlk + bc;
        async_b128(gb,     &Bs[0][br][bc]);
        async_b128(gb + 8, &Bs[0][br][bc + 8]);
    }
#endif

    for (int kt = 0; kt < KT; ++kt) {
        const int k = kt << 5;
#if USE_ASYNC
        wait_async0();
        __syncthreads();
        if (kt + 1 < KT) {
            const int kn = k + 32;
            const __bf16* ga = A + (size_t)(mBlk + ar) * K + kn + ac;
            async_b128(ga, &As[buf ^ 1][ar][ac]);
            const __bf16* gb = Bt + (size_t)(kn + br) * N + nBlk + bc;
            async_b128(gb,     &Bs[buf ^ 1][br][bc]);
            async_b128(gb + 8, &Bs[buf ^ 1][br][bc + 8]);
        }
#else
        __syncthreads();
        {
            const __bf16* ga = A + (size_t)(mBlk + ar) * K + k + ac;
            *(uint4*)&As[0][ar][ac] = *(const uint4*)ga;
            const __bf16* gb = Bt + (size_t)(k + br) * N + nBlk + bc;
            *(uint4*)&Bs[0][br][bc]     = *(const uint4*)gb;
            *(uint4*)&Bs[0][br][bc + 8] = *(const uint4*)(gb + 8);
        }
        __syncthreads();
#endif
        v16bf a[2], b[2];
        // A fragment: lane r holds row; lanes 0-15 -> K {0..7,16..23},
        // lanes 16-31 -> K {8..15,24..31}
#pragma unroll
        for (int i = 0; i < 2; ++i) {
            const __bf16* ap = &As[buf][wm * 32 + i * 16 + r][hb * 8];
            v8bf lo = *(const v8bf*)(ap);
            v8bf hi = *(const v8bf*)(ap + 16);
#pragma unroll
            for (int tt = 0; tt < 8; ++tt) { a[i][tt] = lo[tt]; a[i][8 + tt] = hi[tt]; }
        }
        // B fragment: lane r holds column; lanes 0-15 -> K {0..7,16..23},
        // lanes 16-31 -> K {8..15,24..31}
#pragma unroll
        for (int j = 0; j < 2; ++j) {
            const __bf16* bp = &Bs[buf][hb * 8][wn * 32 + j * 16 + r];
#pragma unroll
            for (int tt = 0; tt < 8; ++tt) { b[j][tt] = bp[tt * BSTR]; b[j][8 + tt] = bp[(16 + tt) * BSTR]; }
        }
#pragma unroll
        for (int i = 0; i < 2; ++i)
#pragma unroll
            for (int j = 0; j < 2; ++j)
                acc[i][j] = __builtin_amdgcn_wmma_f32_16x16x32_bf16(
                    false, a[i], false, b[j], (short)0, acc[i][j], false, false);
#if USE_ASYNC
        buf ^= 1;
#endif
    }

    // C/D layout: lanes 0-15 -> N=lane, rows M=0..7 ; lanes 16-31 -> N=lane-16, rows 8..15
    const int col = lane & 15;
    const int rb  = (lane >> 4) * 8;
#pragma unroll
    for (int i = 0; i < 2; ++i)
#pragma unroll
        for (int j = 0; j < 2; ++j)
#pragma unroll
            for (int tt = 0; tt < 8; ++tt)
                C[(size_t)(mBlk + wm * 32 + i * 16 + rb + tt) * N + nBlk + wn * 32 + j * 16 + col]
                    = acc[i][j][tt];
}

// ============================================================================
// helpers: fp32 -> bf16 conversions / weight transposes
// ============================================================================
__global__ void convert_bf16(const float* __restrict__ src, __bf16* __restrict__ dst, size_t n)
{
    size_t i = (size_t)blockIdx.x * 256 + threadIdx.x;
    if (i < n) dst[i] = (__bf16)src[i];
}

// dst[k*Npad + n] = (n < N) ? src[n*K + k] : 0   (transpose [N,K] -> [K,Npad], bf16)
__global__ void transpose_bf16(const float* __restrict__ src, __bf16* __restrict__ dst,
                               int N, int K, int Npad)
{
    size_t i = (size_t)blockIdx.x * 256 + threadIdx.x;
    if (i >= (size_t)K * Npad) return;
    int n = (int)(i % Npad);
    int k = (int)(i / Npad);
    dst[i] = (n < N) ? (__bf16)src[(size_t)n * K + k] : (__bf16)0.0f;
}

// strided fp32 -> bf16 (extract dt_low columns [0,64) from x_dbl ld=128)
__global__ void convert_strided_bf16(const float* __restrict__ src, __bf16* __restrict__ dst,
                                     int rows, int cols, int lds)
{
    size_t i = (size_t)blockIdx.x * 256 + threadIdx.x;
    if (i >= (size_t)rows * cols) return;
    int r = (int)(i / cols), c = (int)(i % cols);
    dst[i] = (__bf16)src[(size_t)r * lds + c];
}

// ============================================================================
// depthwise causal conv1d (k=4) + bias + SiLU; sliding window per (b,d)
// ============================================================================
__global__ __launch_bounds__(256) void conv_silu(
    const float* __restrict__ xz, const float* __restrict__ w,
    const float* __restrict__ bias, float* __restrict__ u, __bf16* __restrict__ ubf)
{
    int idx = blockIdx.x * 256 + threadIdx.x;   // 0..Bn*DIn-1
    int b = idx >> 11;                           // / DIn
    int d = idx & (DIn - 1);
    float w0 = w[d * 4 + 0], w1 = w[d * 4 + 1], w2 = w[d * 4 + 2], w3 = w[d * 4 + 3];
    float bs = bias[d];
    float x0 = 0.f, x1 = 0.f, x2 = 0.f;
    const float* xp = xz + (size_t)b * Ln * NXZ + d;   // x_in = xz[:, :DIn]
    float*  up  = u   + (size_t)b * Ln * DIn + d;
    __bf16* ubp = ubf + (size_t)b * Ln * DIn + d;
    for (int l = 0; l < Ln; ++l) {
        float x3 = xp[(size_t)l * NXZ];
        float a = fmaf(w0, x0, fmaf(w1, x1, fmaf(w2, x2, fmaf(w3, x3, bs))));
        float s = a / (1.0f + __expf(-a));       // SiLU
        up[(size_t)l * DIn]  = s;
        ubp[(size_t)l * DIn] = (__bf16)s;
        x0 = x1; x1 = x2; x2 = x3;
    }
}

// delta = softplus(dt + 2*b_dt)  (faithful double-bias of the reference)
__global__ void softplus_bias(float* __restrict__ delta, const float* __restrict__ b_dt, size_t n)
{
    size_t i = (size_t)blockIdx.x * 256 + threadIdx.x;
    if (i >= n) return;
    int d = (int)(i & (DIn - 1));
    float x = delta[i] + 2.0f * b_dt[d];
    delta[i] = (x > 20.0f) ? x : log1pf(__expf(x));
}

// ============================================================================
// selective scan: one thread per (b, d, s) state; y reduced over the 16 states
// of a channel with wave32 shfl_xor (lanes [16c,16c+15] share a channel).
// ============================================================================
__global__ __launch_bounds__(256) void scan_kernel(
    const float* __restrict__ delta, const float* __restrict__ u,
    const float* __restrict__ xdbl,  const float* __restrict__ A_log,
    float* __restrict__ yss)
{
    int s    = threadIdx.x & 15;
    int chan = (blockIdx.x * 256 + threadIdx.x) >> 4;   // 0 .. Bn*DIn-1
    int b = chan >> 11;
    int d = chan & (DIn - 1);

    float As = -__expf(A_log[d * DSn + s]);
    float h = 0.0f;
    const float* dp = delta + (size_t)b * Ln * DIn + d;
    const float* up = u     + (size_t)b * Ln * DIn + d;
    const float* xp = xdbl  + (size_t)b * Ln * NXP + DTRn + s;  // B at +64, C at +80
    float* yp = yss + (size_t)b * Ln * DIn + d;

    for (int l = 0; l < Ln; ++l) {
        float dlt = dp[(size_t)l * DIn];
        float duu = dlt * up[(size_t)l * DIn];
        float Bv  = xp[(size_t)l * NXP];
        float Cv  = xp[(size_t)l * NXP + DSn];
        h = fmaf(__expf(dlt * As), h, duu * Bv);
        float y = h * Cv;
        y += __shfl_xor(y, 1);
        y += __shfl_xor(y, 2);
        y += __shfl_xor(y, 4);
        y += __shfl_xor(y, 8);
        if (s == 0) yp[(size_t)l * DIn] = y;
    }
}

// yy = (y_ssm + u*D) * silu(z), emitted as bf16 for the out_proj GEMM
__global__ void finalize_gate(const float* __restrict__ yss, const float* __restrict__ u,
                              const float* __restrict__ Dv, const float* __restrict__ xz,
                              __bf16* __restrict__ yybf, size_t n)
{
    size_t i = (size_t)blockIdx.x * 256 + threadIdx.x;
    if (i >= n) return;
    int d = (int)(i & (DIn - 1));
    size_t row = i >> 11;
    float z  = xz[row * NXZ + DIn + d];
    float y  = yss[i] + u[i] * Dv[d];
    float sz = z / (1.0f + __expf(-z));
    yybf[i] = (__bf16)(y * sz);
}

// ============================================================================
// launcher
// ============================================================================
extern "C" void kernel_launch(void* const* d_in, const int* in_sizes, int n_in,
                              void* d_out, int out_size, void* d_ws, size_t ws_size,
                              hipStream_t stream)
{
    const float* x      = (const float*)d_in[0];
    const float* W_in   = (const float*)d_in[1];
    const float* conv_w = (const float*)d_in[2];
    const float* conv_b = (const float*)d_in[3];
    const float* W_x    = (const float*)d_in[4];
    const float* W_dt   = (const float*)d_in[5];
    const float* b_dt   = (const float*)d_in[6];
    const float* A_log  = (const float*)d_in[7];
    const float* Dv     = (const float*)d_in[8];
    const float* W_out  = (const float*)d_in[9];
    float* out = (float*)d_out;

    // ---- workspace layout (256B aligned) ----
    char* ws = (char*)d_ws;
    size_t off = 0;
    auto alloc = [&](size_t bytes) { void* p = ws + off; off += (bytes + 255) & ~(size_t)255; return p; };
    float*  xz     = (float*) alloc((size_t)Mrows * NXZ * 4);     // 64 MB
    float*  u      = (float*) alloc((size_t)Mrows * DIn * 4);     // 32 MB
    __bf16* ubf    = (__bf16*)alloc((size_t)Mrows * DIn * 2);     // 16 MB
    float*  xdbl   = (float*) alloc((size_t)Mrows * NXP * 4);     //  2 MB
    float*  delta  = (float*) alloc((size_t)Mrows * DIn * 4);     // 32 MB
    float*  yss    = (float*) alloc((size_t)Mrows * DIn * 4);     // 32 MB
    __bf16* xbf    = (__bf16*)alloc((size_t)Mrows * DMn * 2);     //  8 MB
    __bf16* WinT   = (__bf16*)alloc((size_t)DMn * NXZ * 2);       //  8 MB
    __bf16* WxT    = (__bf16*)alloc((size_t)DIn * NXP * 2);       // .5 MB
    __bf16* dtlow  = (__bf16*)alloc((size_t)Mrows * DTRn * 2);    // .5 MB
    __bf16* WdtT   = (__bf16*)alloc((size_t)DTRn * DIn * 2);      // .25MB
    __bf16* yybf   = (__bf16*)alloc((size_t)Mrows * DIn * 2);     // 16 MB
    __bf16* WoutT  = (__bf16*)alloc((size_t)DIn * DMn * 2);       //  4 MB
    (void)ws_size; (void)n_in; (void)in_sizes; (void)out_size;

    auto blks = [](size_t n) { return (unsigned)((n + 255) / 256); };

    // --- precision casts / weight transposes ---
    convert_bf16<<<blks((size_t)Mrows * DMn), 256, 0, stream>>>(x, xbf, (size_t)Mrows * DMn);
    transpose_bf16<<<blks((size_t)DMn * NXZ), 256, 0, stream>>>(W_in,  WinT,  NXZ, DMn, NXZ);
    transpose_bf16<<<blks((size_t)DIn * NXP), 256, 0, stream>>>(W_x,   WxT,   DTRn + 2 * DSn, DIn, NXP);
    transpose_bf16<<<blks((size_t)DTRn * DIn), 256, 0, stream>>>(W_dt, WdtT,  DIn, DTRn, DIn);
    transpose_bf16<<<blks((size_t)DIn * DMn), 256, 0, stream>>>(W_out, WoutT, DMn, DIn, DMn);

    // --- in_proj: xz = x @ W_in^T   [4096 x 4096], K=1024 ---
    gemm_bf16_wmma<<<dim3(NXZ / 128, Mrows / 64), 256, 0, stream>>>(xbf, WinT, xz, Mrows, NXZ, DMn);

    // --- depthwise causal conv + SiLU ---
    conv_silu<<<(Bn * DIn) / 256, 256, 0, stream>>>(xz, conv_w, conv_b, u, ubf);

    // --- x_proj: x_dbl = u @ W_x^T  [4096 x 128(pad)], K=2048 ---
    gemm_bf16_wmma<<<dim3(NXP / 128, Mrows / 64), 256, 0, stream>>>(ubf, WxT, xdbl, Mrows, NXP, DIn);

    // --- dt_proj: delta_raw = dt_low @ W_dt^T  [4096 x 2048], K=64 ---
    convert_strided_bf16<<<blks((size_t)Mrows * DTRn), 256, 0, stream>>>(xdbl, dtlow, Mrows, DTRn, NXP);
    gemm_bf16_wmma<<<dim3(DIn / 128, Mrows / 64), 256, 0, stream>>>(dtlow, WdtT, delta, Mrows, DIn, DTRn);
    softplus_bias<<<blks((size_t)Mrows * DIn), 256, 0, stream>>>(delta, b_dt, (size_t)Mrows * DIn);

    // --- selective scan ---
    scan_kernel<<<(Bn * DIn * DSn) / 256, 256, 0, stream>>>(delta, u, xdbl, A_log, yss);

    // --- gate + skip ---
    finalize_gate<<<blks((size_t)Mrows * DIn), 256, 0, stream>>>(yss, u, Dv, xz, yybf, (size_t)Mrows * DIn);

    // --- out_proj: out = yy @ W_out^T  [4096 x 1024], K=2048 ---
    gemm_bf16_wmma<<<dim3(DMn / 128, Mrows / 64), 256, 0, stream>>>(yybf, WoutT, out, Mrows, DMn, DIn);
}
